// Quant_Residual131_custom_folded_47373489275065
// MI455X (gfx1250) — 16-bit and 8-bit matrix operand order repaired and checked by reading, compiled, not run on hardware
//
#include <hip/hip_runtime.h>
#include <hip/hip_bf16.h>
#include <stdint.h>

typedef __attribute__((ext_vector_type(16))) _Float16 v16h;
typedef __attribute__((ext_vector_type(8)))  float    v8f;
typedef __attribute__((ext_vector_type(8)))  int      v8i;
typedef __attribute__((ext_vector_type(4)))  int      v4i;
typedef __attribute__((ext_vector_type(2)))  int      v2i;

#define N_IMG 32
#define CIN   256
#define CMID  128
#define COUT  512
#define HWD   56
#define NPIX  (HWD*HWD)
#define MTOT  (N_IMG*NPIX)
#define MBLK  128
#define NMBLK (MTOT/MBLK)

static constexpr size_t OFF_SCALES = 0;
static constexpr size_t OFF_B1  = 1024;
static constexpr size_t OFF_B2  = 2048;
static constexpr size_t OFF_B3  = 4096;
static constexpr size_t OFF_W1F = 8192;
static constexpr size_t OFF_W2F = OFF_W1F + 131072;
static constexpr size_t OFF_W3F = OFF_W2F + 589824;
static constexpr size_t OFF_W1P = OFF_W3F + 262144;
static constexpr size_t OFF_W2P = OFF_W1P + 65536;
static constexpr size_t OFF_W3P = OFF_W2P + 147456;
static constexpr size_t OFF_XH  = (OFF_W3P + 65536 + 4095) & ~(size_t)4095;
static constexpr size_t OFF_A1  = OFF_XH + (size_t)MTOT*CIN*2;
static constexpr size_t OFF_A2  = OFF_A1 + (size_t)MTOT*CMID;

__global__ void k_init(unsigned* scales) {
    if (threadIdx.x < 3) scales[threadIdx.x] = 0u;
}

__global__ void k_fold_w(const float* __restrict__ w, const float* __restrict__ g,
                         const float* __restrict__ v, float* __restrict__ wf,
                         unsigned* maxslot, int O, int IK) {
    int i = blockIdx.x * 256 + threadIdx.x;
    if (i >= O * IK) return;
    int o = i / IK;
    float sc = g[o] * rsqrtf(v[o] + 1e-3f);
    float f = w[i] * sc;
    wf[i] = f;
    atomicMax(maxslot, __float_as_uint(fabsf(f)));
}

__global__ void k_fold_b(const float* __restrict__ b, const float* __restrict__ g,
                         const float* __restrict__ be, const float* __restrict__ m,
                         const float* __restrict__ v, float* __restrict__ bf, int O) {
    int o = blockIdx.x * blockDim.x + threadIdx.x;
    if (o >= O) return;
    float sc = g[o] * rsqrtf(v[o] + 1e-3f);
    bf[o] = (b[o] - m[o]) * sc + be[o];
}

__global__ void k_x2nhwc(const float* __restrict__ x, _Float16* __restrict__ xh) {
    int i = blockIdx.x * 256 + threadIdx.x;
    if (i >= MTOT * CIN) return;
    int c  = i % CIN;
    int mp = i / CIN;
    int n = mp / NPIX, p = mp % NPIX;
    xh[i] = (_Float16)x[(size_t)(n * CIN + c) * NPIX + p];
}

__global__ void k_pack_w1(const float* __restrict__ w1f, const unsigned* scales,
                          _Float16* __restrict__ w1p) {
    int tid = blockIdx.x * 256 + threadIdx.x;
    int e    = tid & 15;
    int lane = (tid >> 4) & 31;
    int f    = tid >> 9;
    int ks = f >> 3, nbk = f & 7;
    int hl = lane >> 4, ln = lane & 15;
    int c  = ks * 32 + hl * 16 + e;
    int nn = nbk * 16 + ln;
    float mx = fmaxf(__uint_as_float(scales[0]), 1e-8f);
    float s  = 127.0f / mx;
    float q  = rintf(w1f[nn * CIN + c] * s) / s;
    w1p[(size_t)f * 512 + lane * 16 + e] = (_Float16)q;
}

__global__ void k_pack_w2(const float* __restrict__ w2f, const unsigned* scales,
                          char* __restrict__ w2p) {
    int tid = blockIdx.x * 256 + threadIdx.x;
    int e    = tid & 31;
    int lane = (tid >> 5) & 31;
    int f    = tid >> 10;
    int ks = f >> 3, nbk = f & 7;
    int hl = lane >> 4, ln = lane & 15;
    int vv = e >> 2, bb = e & 3;
    int kl = (vv >> 1) * 16 + hl * 8 + (vv & 1) * 4 + bb;
    int k  = ks * 64 + kl;
    int tap = k >> 7, c = k & 127;
    int ky = tap / 3, kx = tap % 3;
    int nn = nbk * 16 + ln;
    float mx = fmaxf(__uint_as_float(scales[1]), 1e-8f);
    float q  = rintf(w2f[((nn * CMID + c) * 3 + ky) * 3 + kx] * (127.0f / mx));
    q = fminf(fmaxf(q, -127.0f), 127.0f);
    w2p[(size_t)f * 1024 + lane * 32 + e] = (char)(int)q;
}

__global__ void k_pack_w3(const float* __restrict__ w3f, const unsigned* scales,
                          char* __restrict__ w3p) {
    int tid = blockIdx.x * 256 + threadIdx.x;
    int e    = tid & 31;
    int lane = (tid >> 5) & 31;
    int f    = tid >> 10;
    int ks = f >> 5, nbg = f & 31;
    int hl = lane >> 4, ln = lane & 15;
    int vv = e >> 2, bb = e & 3;
    int kl = (vv >> 1) * 16 + hl * 8 + (vv & 1) * 4 + bb;
    int c  = ks * 64 + kl;
    int nn = nbg * 16 + ln;
    float mx = fmaxf(__uint_as_float(scales[2]), 1e-8f);
    float q  = rintf(w3f[nn * CMID + c] * (127.0f / mx));
    q = fminf(fmaxf(q, -127.0f), 127.0f);
    w3p[(size_t)f * 1024 + lane * 32 + e] = (char)(int)q;
}

__device__ __forceinline__ unsigned char qlvl(float y) {
    float t = fminf(fmaxf(y * 25.5f, 0.0f), 255.0f);
    return (unsigned char)__float2int_rn(t);
}

__global__ __launch_bounds__(256) void k_conv1(const _Float16* __restrict__ xh,
                                               const _Float16* __restrict__ w1p,
                                               const float* __restrict__ b1f,
                                               unsigned char* __restrict__ act1) {
    __shared__ char As[MBLK * 64];
    const int t = threadIdx.x;
    const int lane = t & 31, wv = t >> 5;
    const int hl = lane >> 4, ln = lane & 15;
    const int m0 = blockIdx.x * MBLK;
    const int row = t >> 1, seg = t & 1;

    v8f acc[8];
#pragma unroll
    for (int m = 0; m < 8; ++m)
#pragma unroll
        for (int r = 0; r < 8; ++r) acc[m][r] = 0.0f;

    for (int ks = 0; ks < 8; ++ks) {
        {
            const char* src = (const char*)(xh + (size_t)(m0 + row) * CIN + ks * 32) + seg * 32;
            char* dst = As + row * 64 + seg * 32;
            ((v4i*)dst)[0] = ((const v4i*)src)[0];
            ((v4i*)dst)[1] = ((const v4i*)src)[1];
        }
        __syncthreads();
        if (ks < 7)
            __builtin_prefetch((const char*)(xh + (size_t)(m0 + row) * CIN + (ks + 1) * 32) + seg * 32, 0, 3);

        v16h b = *(const v16h*)((const char*)w1p + ((size_t)(ks * 8 + wv) * 1024) + lane * 32);

#pragma unroll
        for (int m = 0; m < 8; ++m) {
            const char* ap = As + (m * 16 + ln) * 64 + hl * 32;
            union { v4i q[2]; v16h h; } u;
            u.q[0] = *(const v4i*)ap;
            u.q[1] = *(const v4i*)(ap + 16);
            acc[m] = __builtin_amdgcn_wmma_f32_16x16x32_f16(false, u.h, false, b,
                                                            (short)0, acc[m], false, false);
        }
        __syncthreads();
    }

    const int co = wv * 16 + ln;
    const float bias = b1f[co];
#pragma unroll
    for (int m = 0; m < 8; ++m)
#pragma unroll
        for (int r = 0; r < 8; ++r) {
            int M = m0 + m * 16 + hl * 8 + r;
            act1[(size_t)M * CMID + co] = qlvl(acc[m][r] + bias);
        }
}

__global__ __launch_bounds__(256) void k_conv2(const unsigned char* __restrict__ act1,
                                               const char* __restrict__ w2p,
                                               const float* __restrict__ b2f,
                                               const unsigned* __restrict__ scales,
                                               unsigned char* __restrict__ act2) {
    __shared__ char As[MBLK * 64];
    const int t = threadIdx.x;
    const int lane = t & 31, wv = t >> 5;
    const int hl = lane >> 4, ln = lane & 15;
    const int m0 = blockIdx.x * MBLK;
    const int row = t >> 1, seg = t & 1;
    const int Mr = m0 + row;
    const int n = Mr / NPIX, p = Mr % NPIX;
    const int h = p / HWD, w = p % HWD;

    v8i acc[8];
#pragma unroll
    for (int m = 0; m < 8; ++m)
#pragma unroll
        for (int r = 0; r < 8; ++r) acc[m][r] = 0;

    for (int ks = 0; ks < 18; ++ks) {
        {
            const int tap = ks >> 1, ch = (ks & 1) * 64;
            const int hy = h + tap / 3 - 1, wx = w + tap % 3 - 1;
            char* dst = As + row * 64 + seg * 32;
            if ((unsigned)hy < HWD && (unsigned)wx < HWD) {
                const char* src = (const char*)act1 +
                    ((size_t)(n * NPIX + hy * HWD + wx) * CMID + ch + seg * 32);
                ((v4i*)dst)[0] = ((const v4i*)src)[0];
                ((v4i*)dst)[1] = ((const v4i*)src)[1];
            } else {
                v4i z = {0, 0, 0, 0};
                ((v4i*)dst)[0] = z; ((v4i*)dst)[1] = z;
            }
        }
        __syncthreads();

        v8i b = *(const v8i*)(w2p + ((size_t)(ks * 8 + wv) * 1024) + lane * 32);

#pragma unroll
        for (int m = 0; m < 8; ++m) {
            const char* ap = As + (m * 16 + ln) * 64 + hl * 8;
            union { v2i q[4]; v8i a; } u;
            u.q[0] = *(const v2i*)(ap);
            u.q[1] = *(const v2i*)(ap + 16);
            u.q[2] = *(const v2i*)(ap + 32);
            u.q[3] = *(const v2i*)(ap + 48);
            acc[m] = __builtin_amdgcn_wmma_i32_16x16x64_iu8(false, u.a, true, b,
                                                            acc[m], false, false);
        }
        __syncthreads();
    }

    const int co = wv * 16 + ln;
    const float bias = b2f[co];
    const float sc = __uint_as_float(scales[1]) / (25.5f * 127.0f);
#pragma unroll
    for (int m = 0; m < 8; ++m)
#pragma unroll
        for (int r = 0; r < 8; ++r) {
            int M = m0 + m * 16 + hl * 8 + r;
            act2[(size_t)M * CMID + co] = qlvl((float)acc[m][r] * sc + bias);
        }
}

__global__ __launch_bounds__(256) void k_conv3(const unsigned char* __restrict__ act2,
                                               const char* __restrict__ w3p,
                                               const float* __restrict__ b3f,
                                               const unsigned* __restrict__ scales,
                                               float* __restrict__ out) {
    __shared__ char As[MBLK * 64];
    const int t = threadIdx.x;
    const int lane = t & 31, wv = t >> 5;
    const int hl = lane >> 4, ln = lane & 15;
    const int m0 = blockIdx.x * MBLK;
    const int nbg = blockIdx.y * 8 + wv;
    const int row = t >> 1, seg = t & 1;

    v8i acc[8];
#pragma unroll
    for (int m = 0; m < 8; ++m)
#pragma unroll
        for (int r = 0; r < 8; ++r) acc[m][r] = 0;

    for (int ks = 0; ks < 2; ++ks) {
        {
            const char* src = (const char*)act2 + ((size_t)(m0 + row) * CMID + ks * 64 + seg * 32);
            char* dst = As + row * 64 + seg * 32;
            ((v4i*)dst)[0] = ((const v4i*)src)[0];
            ((v4i*)dst)[1] = ((const v4i*)src)[1];
        }
        __syncthreads();

        v8i b = *(const v8i*)(w3p + ((size_t)(ks * 32 + nbg) * 1024) + lane * 32);

#pragma unroll
        for (int m = 0; m < 8; ++m) {
            const char* ap = As + (m * 16 + ln) * 64 + hl * 8;
            union { v2i q[4]; v8i a; } u;
            u.q[0] = *(const v2i*)(ap);
            u.q[1] = *(const v2i*)(ap + 16);
            u.q[2] = *(const v2i*)(ap + 32);
            u.q[3] = *(const v2i*)(ap + 48);
            acc[m] = __builtin_amdgcn_wmma_i32_16x16x64_iu8(false, u.a, true, b,
                                                            acc[m], false, false);
        }
        __syncthreads();
    }

    const int co = nbg * 16 + ln;
    const float bias = b3f[co];
    const float sc = __uint_as_float(scales[2]) / (25.5f * 127.0f);
#pragma unroll
    for (int m = 0; m < 8; ++m)
#pragma unroll
        for (int r = 0; r < 8; ++r) {
            int M = m0 + m * 16 + hl * 8 + r;
            int nn = M / NPIX, pp = M % NPIX;
            out[(size_t)(nn * COUT + co) * NPIX + pp] = (float)acc[m][r] * sc + bias;
        }
}

extern "C" void kernel_launch(void* const* d_in, const int* in_sizes, int n_in,
                              void* d_out, int out_size, void* d_ws, size_t ws_size,
                              hipStream_t stream) {
    (void)in_sizes; (void)n_in; (void)out_size; (void)ws_size;
    const float* x  = (const float*)d_in[0];
    const float* w1 = (const float*)d_in[1];
    const float* b1 = (const float*)d_in[2];
    const float* g1 = (const float*)d_in[3];
    const float* be1= (const float*)d_in[4];
    const float* m1 = (const float*)d_in[5];
    const float* v1 = (const float*)d_in[6];
    const float* w2 = (const float*)d_in[7];
    const float* b2 = (const float*)d_in[8];
    const float* g2 = (const float*)d_in[9];
    const float* be2= (const float*)d_in[10];
    const float* m2 = (const float*)d_in[11];
    const float* v2 = (const float*)d_in[12];
    const float* w3 = (const float*)d_in[13];
    const float* b3 = (const float*)d_in[14];
    const float* g3 = (const float*)d_in[15];
    const float* be3= (const float*)d_in[16];
    const float* m3 = (const float*)d_in[17];
    const float* v3 = (const float*)d_in[18];
    float* out = (float*)d_out;

    char* ws = (char*)d_ws;
    unsigned* scales   = (unsigned*)(ws + OFF_SCALES);
    float* b1f         = (float*)(ws + OFF_B1);
    float* b2f         = (float*)(ws + OFF_B2);
    float* b3f         = (float*)(ws + OFF_B3);
    float* w1f         = (float*)(ws + OFF_W1F);
    float* w2f         = (float*)(ws + OFF_W2F);
    float* w3f         = (float*)(ws + OFF_W3F);
    _Float16* w1p      = (_Float16*)(ws + OFF_W1P);
    char* w2p          = (char*)(ws + OFF_W2P);
    char* w3p          = (char*)(ws + OFF_W3P);
    _Float16* xh       = (_Float16*)(ws + OFF_XH);
    unsigned char* a1  = (unsigned char*)(ws + OFF_A1);
    unsigned char* a2  = (unsigned char*)(ws + OFF_A2);

    k_init<<<1, 32, 0, stream>>>(scales);

    k_fold_w<<<(32768 + 255) / 256, 256, 0, stream>>>(w1, g1, v1, w1f, scales + 0, 128, 256);
    k_fold_w<<<(147456 + 255) / 256, 256, 0, stream>>>(w2, g2, v2, w2f, scales + 1, 128, 1152);
    k_fold_w<<<(65536 + 255) / 256, 256, 0, stream>>>(w3, g3, v3, w3f, scales + 2, 512, 128);
    k_fold_b<<<1, 128, 0, stream>>>(b1, g1, be1, m1, v1, b1f, 128);
    k_fold_b<<<1, 128, 0, stream>>>(b2, g2, be2, m2, v2, b2f, 128);
    k_fold_b<<<2, 256, 0, stream>>>(b3, g3, be3, m3, v3, b3f, 512);

    k_pack_w1<<<32768 / 256, 256, 0, stream>>>(w1f, scales, w1p);
    k_pack_w2<<<147456 / 256, 256, 0, stream>>>(w2f, scales, w2p);
    k_pack_w3<<<65536 / 256, 256, 0, stream>>>(w3f, scales, w3p);

    k_x2nhwc<<<((size_t)MTOT * CIN + 255) / 256, 256, 0, stream>>>(x, xh);

    k_conv1<<<NMBLK, 256, 0, stream>>>(xh, w1p, b1f, a1);
    k_conv2<<<NMBLK, 256, 0, stream>>>(a1, w2p, b2f, scales, a2);
    k_conv3<<<dim3(NMBLK, 4), 256, 0, stream>>>(a2, w3p, b3f, scales, out);
}
